// MultiHeadAttention_36962488549863
// MI455X (gfx1250) — matrix operand order repaired and checked by reading, not compiled, not run on hardware
//
#include <hip/hip_runtime.h>
#include <hip/hip_bf16.h>

// ---------------- problem constants ----------------
constexpr int D_MODEL = 768;
constexpr int N_HEADS = 12;
constexpr int D_KH    = 64;          // head dim
constexpr int SEQLEN  = 2048;
constexpr int NBATCH  = 4;
constexpr int MTOT    = NBATCH * SEQLEN;   // 8192
constexpr int NQKV    = 3 * D_MODEL;       // 2304

// ---------------- WMMA types (gfx1250, wave32) ----------------
typedef __attribute__((ext_vector_type(16))) __bf16 v16bf;
typedef __attribute__((ext_vector_type(8)))  float  v8f;
typedef __attribute__((ext_vector_type(4)))  unsigned int v4u;
typedef __attribute__((ext_vector_type(8)))  int v8i;
typedef __attribute__((ext_vector_type(4)))  int v4i;

struct alignas(16) U4 { unsigned int x, y, z, w; };
union FragAB {                 // 16 bf16 = 32 bytes = one WMMA A/B fragment
    U4 u[2];
    v16bf v;
    unsigned short s[16];
};
union U4S { U4 q; unsigned short s[8]; };

__device__ __forceinline__ unsigned short f2bf(float f) {
    unsigned int u = __builtin_bit_cast(unsigned int, f);
    unsigned int r = u + 0x7FFFu + ((u >> 16) & 1u);   // round-to-nearest-even
    return (unsigned short)(r >> 16);
}

__device__ __forceinline__ v8f wmma_bf16(v16bf a, v16bf b, v8f c) {
    // D(16x16,f32) = A(16x32,bf16) * B(32x16,bf16) + C  -> v_wmma_f32_16x16x32_bf16
    return __builtin_amdgcn_wmma_f32_16x16x32_bf16(false, a, false, b,
                                                   (short)0, c, false, false);
}

// ---------------- TDM: async 2D bf16 tile load Global -> LDS ----------------
// Builds a Tensor DMA Descriptor (D#) per CDNA5 ISA
// and issues TENSOR_LOAD_TO_LDS (tracked by TENSORcnt). data_size = 2 bytes.
// pad_interval_code: pad after 2<<code DWORDs; pad_amount_code: (code+1) DWORDs.
// Toolchain here is the 6-arg builtin flavor (clang-23 / therock-10.0 headers).
__device__ __forceinline__ void tdm_load_2d_bf16(unsigned lds_off, const void* gaddr,
                                                 unsigned dim0, unsigned dim1,
                                                 unsigned stride0,
                                                 unsigned tile0, unsigned tile1,
                                                 unsigned pad_interval_code,
                                                 unsigned pad_amount_code) {
    unsigned long long ga = (unsigned long long)(uintptr_t)gaddr;
    v4u g0;
    g0[0] = 1u;                                            // count=1, user descriptor
    g0[1] = lds_off;                                       // LDS byte address
    g0[2] = (unsigned)(ga & 0xFFFFFFFFu);                  // global_addr[31:0]
    g0[3] = (unsigned)((ga >> 32) & 0x01FFFFFFu)           // global_addr[56:32]
          | (2u << 30);                                    // type = 2 ("image")
    v8i g1;
    unsigned c0 = (1u << 16)                               // data_size = 2 bytes
                | (1u << 20)                               // pad_enable
                | (pad_interval_code << 22)
                | (pad_amount_code << 25);
    g1[0] = (int)c0;                                       // wg_mask=0 (not in cluster)
    g1[1] = (int)((dim0 & 0xFFFFu) << 16);                 // tensor_dim0[15:0]
    g1[2] = (int)((dim0 >> 16) | ((dim1 & 0xFFFFu) << 16));// dim0[31:16] | dim1[15:0]
    g1[3] = (int)((dim1 >> 16) | (tile0 << 16));           // dim1[31:16] | tile_dim0
    g1[4] = (int)(tile1 & 0xFFFFu);                        // tile_dim1 | tile_dim2=0
    g1[5] = (int)stride0;                                  // tensor_dim0_stride[31:0]
    g1[6] = 0;                                             // stride0[47:32] | stride1 lo
    g1[7] = 0;
    v4i z; z[0] = 0; z[1] = 0; z[2] = 0; z[3] = 0;         // 2D: groups 2/3 unused
    v8i z8; z8[0]=0; z8[1]=0; z8[2]=0; z8[3]=0; z8[4]=0; z8[5]=0; z8[6]=0; z8[7]=0;
    __builtin_amdgcn_tensor_load_to_lds(g0, g1, z, z, z8, 0);
}

__device__ __forceinline__ unsigned lds_off_of(const void* p) {
    // generic LDS address: low 32 bits are the LDS byte offset (ISA §10.2)
    return (unsigned)(uintptr_t)p;
}

// ---------------- fp32 -> bf16 conversion ----------------
__global__ __launch_bounds__(256)
void cvt_bf16(const float* __restrict__ s, unsigned short* __restrict__ d, int n4) {
    int i = blockIdx.x * blockDim.x + threadIdx.x;
    if (i < n4) {
        float4 f = ((const float4*)s)[i];
        unsigned long long lo = (unsigned long long)(f2bf(f.x) | ((unsigned int)f2bf(f.y) << 16));
        unsigned long long hi = (unsigned long long)(f2bf(f.z) | ((unsigned int)f2bf(f.w) << 16));
        ((unsigned long long*)d)[i] = lo | (hi << 32);
    }
}

// ---------------- tiled bf16 GEMM: C(M,N) = A(M,K) * W(K,N) + bias ----------------
// MODE 0: fp32 output to outF[M*N]
// MODE 1: QKV epilogue -> scatter bf16 into Q/K/V [B,H,S,64]
template <int MODE>
__global__ __launch_bounds__(256)
void gemm_bf16(const unsigned short* __restrict__ A,
               const unsigned short* __restrict__ W,
               const float* __restrict__ bias,
               float* __restrict__ outF,
               unsigned short* __restrict__ Qd,
               unsigned short* __restrict__ Kd,
               unsigned short* __restrict__ Vd,
               int M, int N, int K) {
    __shared__ alignas(16) unsigned short As[128][40];   // [m][k], k-step 32, pad->40
    __shared__ alignas(16) unsigned short Bs[128][40];   // [n][k] (W transposed in LDS)

    const int tid   = threadIdx.x;
    const int wid   = tid >> 5;           // 8 waves
    const int lane  = tid & 31;
    const int row16 = lane & 15;
    const int grp   = lane >> 4;          // 0 or 1
    const int wm    = wid >> 1;           // 0..3  -> 32 rows each
    const int wn    = wid & 1;            // 0..1  -> 64 cols each
    const int rm    = wm * 32;
    const int rn    = wn * 64;
    const int m0    = blockIdx.y * 128;
    const int n0    = blockIdx.x * 128;
    const unsigned as_lds = lds_off_of(&As[0][0]);

    v8f acc[2][4];
#pragma unroll
    for (int mt = 0; mt < 2; ++mt)
#pragma unroll
        for (int nt = 0; nt < 4; ++nt)
#pragma unroll
            for (int j = 0; j < 8; ++j) acc[mt][nt][j] = 0.f;

    for (int k0 = 0; k0 < K; k0 += 32) {
        __syncthreads();
        // stage A tile 128x32 via Tensor Data Mover (async, padded rows 32->40)
        if (wid == 0) {
            // row = 32 bf16 = 16 DWORDs -> pad_interval code 3; pad 4 DWORDs -> code 3
            tdm_load_2d_bf16(as_lds, A + (size_t)m0 * K + k0,
                             /*dim0*/32, /*dim1*/128, /*stride0*/(unsigned)K,
                             /*tile0*/32, /*tile1*/128, 3, 3);
        }
        // stage W tile 32x128 transposed -> Bs[n][k] (needs transpose: VGPR path)
#pragma unroll
        for (int it = 0; it < 2; ++it) {
            int idx = tid + it * 256;
            int kr = idx >> 4, c8 = idx & 15;
            U4S d;
            d.q = *(const U4*)(W + (size_t)(k0 + kr) * N + n0 + c8 * 8);
#pragma unroll
            for (int e = 0; e < 8; ++e) Bs[c8 * 8 + e][kr] = d.s[e];
        }
        if (wid == 0) __builtin_amdgcn_s_wait_tensorcnt(0);
        __syncthreads();

        FragAB af[2], bfr[4];
#pragma unroll
        for (int mt = 0; mt < 2; ++mt) {
            int r = rm + mt * 16 + row16;
            af[mt].u[0] = *(const U4*)&As[r][grp * 8];        // K = grp*8 .. +7
            af[mt].u[1] = *(const U4*)&As[r][16 + grp * 8];   // K = 16+grp*8 .. +7
        }
#pragma unroll
        for (int nt = 0; nt < 4; ++nt) {
            int n = rn + nt * 16 + row16;
            bfr[nt].u[0] = *(const U4*)&Bs[n][grp * 8];       // K = grp*8 .. +7
            bfr[nt].u[1] = *(const U4*)&Bs[n][16 + grp * 8];  // K = 16+grp*8 .. +7
        }
#pragma unroll
        for (int mt = 0; mt < 2; ++mt)
#pragma unroll
            for (int nt = 0; nt < 4; ++nt)
                acc[mt][nt] = wmma_bf16(af[mt].v, bfr[nt].v, acc[mt][nt]);
    }

    // epilogue (C layout: VGPR j, lanes 0-15 -> M=j, lanes 16-31 -> M=j+8)
#pragma unroll
    for (int mt = 0; mt < 2; ++mt)
#pragma unroll
        for (int nt = 0; nt < 4; ++nt)
#pragma unroll
            for (int j = 0; j < 8; ++j) {
                int r = m0 + rm + mt * 16 + j + 8 * grp;
                int c = n0 + rn + nt * 16 + row16;
                float v = acc[mt][nt][j] + bias[c];
                if (MODE == 0) {
                    outF[(size_t)r * N + c] = v;
                } else {
                    int b = r / SEQLEN, s = r % SEQLEN;
                    int head  = c / (3 * D_KH);
                    int w     = c % (3 * D_KH);
                    int which = w / D_KH;
                    int dk    = w % D_KH;
                    unsigned short* dst = (which == 0) ? Qd : (which == 1) ? Kd : Vd;
                    dst[(((size_t)b * N_HEADS + head) * SEQLEN + s) * D_KH + dk] = f2bf(v);
                }
            }
}

// ---------------- flash attention: per (b,h), 64 query rows / block ----------------
__global__ __launch_bounds__(128)
void flash_attn(const unsigned short* __restrict__ Qm,
                const unsigned short* __restrict__ Km,
                const unsigned short* __restrict__ Vm,
                unsigned short* __restrict__ Ctx) {
    __shared__ alignas(16) unsigned short Ks[64][72];      // [key][dk]  (B-layout for Q*K^T)
    __shared__ alignas(16) unsigned short Vt[64][72];      // [dk][key]  (B-layout for P*V)
    __shared__ alignas(16) unsigned short Ps[4][16][72];   // per-wave P tile, [row][key]

    const int tid   = threadIdx.x;
    const int wid   = tid >> 5;       // 4 waves, 16 query rows each
    const int lane  = tid & 31;
    const int row16 = lane & 15;
    const int grp   = lane >> 4;

    const int bh = blockIdx.y;
    const size_t base = (size_t)bh * SEQLEN * D_KH;
    const unsigned short* Qb = Qm + base;
    const unsigned short* Kb = Km + base;
    const unsigned short* Vb = Vm + base;
    const unsigned ks_lds = lds_off_of(&Ks[0][0]);

    const int q0 = blockIdx.x * 64 + wid * 16;

    // Q fragments (A-layout), loaded once: K-chunks ks=0,1 over d_k=64
    FragAB qf[2];
#pragma unroll
    for (int ks = 0; ks < 2; ++ks) {
        const unsigned short* qp = Qb + (size_t)(q0 + row16) * D_KH + ks * 32 + grp * 8;
        qf[ks].u[0] = *(const U4*)qp;
        qf[ks].u[1] = *(const U4*)(qp + 16);
    }

    v8f o[4];
    float m[8], l[8];
#pragma unroll
    for (int d = 0; d < 4; ++d)
#pragma unroll
        for (int j = 0; j < 8; ++j) o[d][j] = 0.f;
#pragma unroll
    for (int j = 0; j < 8; ++j) { m[j] = -1e30f; l[j] = 0.f; }

    const float scale = 0.125f;  // 1/sqrt(64)

    for (int kc = 0; kc < SEQLEN; kc += 64) {
        __syncthreads();
        // K chunk (64x64 bf16, contiguous) via TDM, rows padded 64->72 shorts:
        // row = 64 bf16 = 32 DWORDs -> pad_interval code 4; pad 4 DWORDs -> code 3
        if (wid == 0) {
            tdm_load_2d_bf16(ks_lds, Kb + (size_t)kc * D_KH,
                             /*dim0*/64, /*dim1*/64, /*stride0*/64,
                             /*tile0*/64, /*tile1*/64, 4, 3);
        }
        // V chunk transposed -> Vt[dk][key] (needs transpose: VGPR path)
#pragma unroll
        for (int it = 0; it < 4; ++it) {
            int idx = tid + it * 128;
            int key = idx >> 3, c8 = idx & 7;
            U4S dv;
            dv.q = *(const U4*)(Vb + (size_t)(kc + key) * D_KH + c8 * 8);
#pragma unroll
            for (int e = 0; e < 8; ++e) Vt[c8 * 8 + e][key] = dv.s[e];
        }
        if (wid == 0) __builtin_amdgcn_s_wait_tensorcnt(0);
        __syncthreads();

        // scores S(16x64) = Q(16x64) * K^T
        v8f s[4];
#pragma unroll
        for (int nt = 0; nt < 4; ++nt) {
#pragma unroll
            for (int j = 0; j < 8; ++j) s[nt][j] = 0.f;
#pragma unroll
            for (int ks = 0; ks < 2; ++ks) {
                FragAB kf;
                kf.u[0] = *(const U4*)&Ks[nt * 16 + row16][ks * 32 + grp * 8];
                kf.u[1] = *(const U4*)&Ks[nt * 16 + row16][ks * 32 + 16 + grp * 8];
                s[nt] = wmma_bf16(qf[ks].v, kf.v, s[nt]);
            }
#pragma unroll
            for (int j = 0; j < 8; ++j) s[nt][j] *= scale;
        }

        // online softmax per row (row = j + 8*grp; reduce over 16-lane half)
#pragma unroll
        for (int j = 0; j < 8; ++j) {
            float t = fmaxf(fmaxf(s[0][j], s[1][j]), fmaxf(s[2][j], s[3][j]));
#pragma unroll
            for (int msk = 8; msk; msk >>= 1) t = fmaxf(t, __shfl_xor(t, msk, 32));
            float mn   = fmaxf(m[j], t);
            float corr = __expf(m[j] - mn);
            m[j] = mn;
            float rs = 0.f;
#pragma unroll
            for (int nt = 0; nt < 4; ++nt) {
                float p = __expf(s[nt][j] - mn);
                s[nt][j] = p;
                rs += p;
            }
#pragma unroll
            for (int msk = 8; msk; msk >>= 1) rs += __shfl_xor(rs, msk, 32);
            l[j] = l[j] * corr + rs;
#pragma unroll
            for (int d = 0; d < 4; ++d) o[d][j] *= corr;
        }

        // C-layout -> A-layout via per-wave LDS round trip
#pragma unroll
        for (int nt = 0; nt < 4; ++nt)
#pragma unroll
            for (int j = 0; j < 8; ++j)
                Ps[wid][j + 8 * grp][nt * 16 + row16] = f2bf(s[nt][j]);
        __syncthreads();

        // O += P(16x64) * V(64x64)
#pragma unroll
        for (int d = 0; d < 4; ++d) {
#pragma unroll
            for (int ks = 0; ks < 2; ++ks) {
                FragAB pf, vf;
                const unsigned short* pp = &Ps[wid][row16][ks * 32 + grp * 8];
                pf.u[0] = *(const U4*)pp;
                pf.u[1] = *(const U4*)(pp + 16);
                vf.u[0] = *(const U4*)&Vt[d * 16 + row16][ks * 32 + grp * 8];
                vf.u[1] = *(const U4*)&Vt[d * 16 + row16][ks * 32 + 16 + grp * 8];
                o[d] = wmma_bf16(pf.v, vf.v, o[d]);
            }
        }
    }

    // normalize & store context bf16 [B*S, D_MODEL], col = head*64 + dk
    const int b = bh / N_HEADS, head = bh % N_HEADS;
#pragma unroll
    for (int d = 0; d < 4; ++d)
#pragma unroll
        for (int j = 0; j < 8; ++j) {
            int sr = q0 + j + 8 * grp;
            float val = o[d][j] / l[j];
            Ctx[(size_t)(b * SEQLEN + sr) * D_MODEL + head * D_KH + d * 16 + row16] = f2bf(val);
        }
}

// ---------------- host glue ----------------
extern "C" void kernel_launch(void* const* d_in, const int* in_sizes, int n_in,
                              void* d_out, int out_size, void* d_ws, size_t ws_size,
                              hipStream_t stream) {
    const float* x    = (const float*)d_in[0];
    const float* Wqkv = (const float*)d_in[1];
    const float* bqkv = (const float*)d_in[2];
    const float* Wout = (const float*)d_in[3];
    const float* bout = (const float*)d_in[4];
    float* out = (float*)d_out;

    // workspace carve (~68 MB total, 256B aligned)
    size_t off = 0;
    auto carve = [&](size_t bytes) -> void* {
        void* p = (char*)d_ws + off;
        off += (bytes + 255) & ~(size_t)255;
        return p;
    };
    unsigned short* Xb    = (unsigned short*)carve((size_t)MTOT * D_MODEL * 2);
    unsigned short* Wqkvb = (unsigned short*)carve((size_t)D_MODEL * NQKV * 2);
    unsigned short* Woutb = (unsigned short*)carve((size_t)D_MODEL * D_MODEL * 2);
    unsigned short* Qd    = (unsigned short*)carve((size_t)MTOT * D_KH * N_HEADS * 2);
    unsigned short* Kd    = (unsigned short*)carve((size_t)MTOT * D_KH * N_HEADS * 2);
    unsigned short* Vd    = (unsigned short*)carve((size_t)MTOT * D_KH * N_HEADS * 2);
    unsigned short* Ctxb  = (unsigned short*)carve((size_t)MTOT * D_MODEL * 2);
    (void)ws_size; (void)in_sizes; (void)n_in; (void)out_size;

    // 1) fp32 -> bf16
    {
        int n4 = MTOT * D_MODEL / 4;
        cvt_bf16<<<(n4 + 255) / 256, 256, 0, stream>>>(x, Xb, n4);
        n4 = D_MODEL * NQKV / 4;
        cvt_bf16<<<(n4 + 255) / 256, 256, 0, stream>>>(Wqkv, Wqkvb, n4);
        n4 = D_MODEL * D_MODEL / 4;
        cvt_bf16<<<(n4 + 255) / 256, 256, 0, stream>>>(Wout, Woutb, n4);
    }

    // 2) fused QKV projection -> Q/K/V [B,H,S,64] bf16
    gemm_bf16<1><<<dim3(NQKV / 128, MTOT / 128), 256, 0, stream>>>(
        Xb, Wqkvb, bqkv, nullptr, Qd, Kd, Vd, MTOT, NQKV, D_MODEL);

    // 3) flash attention -> context bf16 [B*S, D_MODEL]
    flash_attn<<<dim3(SEQLEN / 64, NBATCH * N_HEADS), 128, 0, stream>>>(Qd, Kd, Vd, Ctxb);

    // 4) output projection -> fp32 d_out
    gemm_bf16<0><<<dim3(D_MODEL / 128, MTOT / 128), 256, 0, stream>>>(
        Ctxb, Woutb, bout, out, nullptr, nullptr, nullptr, MTOT, D_MODEL, D_MODEL);
}
